// ScaledDotProductAttention_70858370449467
// MI455X (gfx1250) — matrix operand order repaired and checked by reading, not compiled, not run on hardware
//
#include <hip/hip_runtime.h>
#include <hip/hip_bf16.h>
#include <stdint.h>

// ---------------------------------------------------------------------------
// ScaledDotProductAttention for MI455X (gfx1250, wave32, WMMA)
//   B=8 H=8 Lq=Lk=512 R=4 D=Dv=64
//   scores = (Q/8) @ (sum_r K)^T ; attn = softmax(mask? -1e9 : scores*omega)
//   out    = attn @ (sum_r V)
//   d_out  = [ out (B,H,Lq,64) | attn (B,H,Lq,Lk) ]  (f32, concatenated)
// ---------------------------------------------------------------------------

typedef __attribute__((ext_vector_type(16))) _Float16 v16h;
typedef __attribute__((ext_vector_type(8)))  float    v8f;
typedef __attribute__((ext_vector_type(4)))  float    v4f;

#define TEMP_INV  0.125f
#define NEG_INF_F (-1000000000.0f)

constexpr int Bc = 8, Hc = 8, LQ = 512, LK = 512, Rc = 4, Dc = 64, DV = 64;
constexpr int BH = Bc * Hc;

// ---------------------------------------------------------------------------
// Kernel 1: fold the r-dimension out of k and v, store as f16.
//   ksum  : (BH, LK, D)  row-major  (j-major, d contiguous)  -> B tiles, GEMM1
//   vsumT : (BH, D, LK)  d-major, j contiguous               -> B tiles, GEMM2
// ---------------------------------------------------------------------------
__global__ void kv_reduce_kernel(const float* __restrict__ k,
                                 const float* __restrict__ v,
                                 _Float16* __restrict__ ksum,
                                 _Float16* __restrict__ vsumT)
{
    int idx = blockIdx.x * blockDim.x + threadIdx.x;       // over BH*LK*D
    if (idx >= BH * LK * Dc) return;
    int d  = idx & (Dc - 1);
    int j  = (idx >> 6) & (LK - 1);
    int bh = idx >> 15;

    const float* kp = k + ((size_t)bh * LK + j) * Rc * Dc + d;
    const float* vp = v + ((size_t)bh * LK + j) * Rc * DV + d;
    float ks = 0.f, vs = 0.f;
#pragma unroll
    for (int r = 0; r < Rc; ++r) { ks += kp[r * Dc]; vs += vp[r * DV]; }

    ksum[idx] = (_Float16)ks;
    vsumT[((size_t)bh * Dc + d) * LK + j] = (_Float16)vs;
}

// ---------------------------------------------------------------------------
// Kernel 2: one wave32 per (bh, 16-row q tile). grid.x = BH * (LQ/16) = 2048.
// ---------------------------------------------------------------------------
__global__ void
__launch_bounds__(32)
attn_wmma_kernel(const float* __restrict__ q,
                 const float* __restrict__ omega,
                 const unsigned char* __restrict__ mask,   // jnp bool: 1 byte
                 const _Float16* __restrict__ ksum,
                 const _Float16* __restrict__ vsumT,
                 float* __restrict__ out,                  // (BH, LQ, 64)
                 float* __restrict__ attn)                 // (BH, LQ, LK)
{
    constexpr int PAD = 516;                 // 516 % 64 == 4 -> bank spread
    __shared__ float sc[16][PAD];            // 16 x 512 score strip (+pad)
    __shared__ float rowinv[16];             // 1/rowsum broadcast

    const int bh = blockIdx.x >> 5;          // (b,h)
    const int qt = blockIdx.x & 31;
    const int q0 = qt * 16;
    const int b  = bh >> 3;                  // H == 8

    const int lane = threadIdx.x;            // wave32: 0..31, EXEC all ones
    const int half = lane >> 4;              // 0: lanes 0-15, 1: lanes 16-31
    const int l16  = lane & 15;

    // ---- A tiles from Q (16x64, scaled by 1/TEMPERATURE), f16, contiguous k order ----
    // lane: M = l16; elem e (0..15) -> K = base + half*16 + e
    const float* qrow = q + ((size_t)bh * LQ + q0 + l16) * Dc;
    v16h a0, a1;
#pragma unroll
    for (int e = 0; e < 8; ++e) {
        a0[e]     = (_Float16)(qrow[      half * 16 + e] * TEMP_INV);
        a0[e + 8] = (_Float16)(qrow[8   + half * 16 + e] * TEMP_INV);
        a1[e]     = (_Float16)(qrow[32  + half * 16 + e] * TEMP_INV);
        a1[e + 8] = (_Float16)(qrow[40  + half * 16 + e] * TEMP_INV);
    }

    // ---- GEMM1: scores = A @ Ksum^T, 32 column tiles of 16 ----------------
    const _Float16* kb = ksum + (size_t)bh * LK * Dc;
    float m[8];
#pragma unroll
    for (int r = 0; r < 8; ++r) m[r] = -INFINITY;

    for (int jt = 0; jt < 32; ++jt) {
        const int j0  = jt * 16;
        const int col = j0 + l16;
        // B 32x16 f16, same k order as A: lane N = l16, elem e -> K = half*16 + e
        const _Float16* kcol = kb + (size_t)col * Dc;          // row j, d-contig
        v16h b0 = *(const v16h*)(kcol +      half * 16);       // d = 0..31
        v16h b1 = *(const v16h*)(kcol + 32 + half * 16);       // d = 32..63
        if (jt < 31)                                           // next column tile
            __builtin_prefetch((const void*)(kcol + 16 * Dc), 0, 3);

        v8f c = {};
        c = __builtin_amdgcn_wmma_f32_16x16x32_f16(false, a0, false, b0,
                                                   (short)0, c, false, false);
        c = __builtin_amdgcn_wmma_f32_16x16x32_f16(false, a1, false, b1,
                                                   (short)0, c, false, false);

        // C layout: VGPR r, lanes 0-15 -> row r, lanes 16-31 -> row r+8
#pragma unroll
        for (int r = 0; r < 8; ++r) {
            const int row = r + half * 8;
            const size_t gi = ((size_t)b * LQ + q0 + row) * LK + col;
            float s = c[r] * omega[gi];
            if (mask[gi]) s = NEG_INF_F;
            m[r] = fmaxf(m[r], s);
            sc[row][col] = s;
        }
    }
    __syncthreads();

    // ---- row max across the 16 lanes of each half ------------------------
#pragma unroll
    for (int r = 0; r < 8; ++r) {
        m[r] = fmaxf(m[r], __shfl_xor(m[r], 1, 32));
        m[r] = fmaxf(m[r], __shfl_xor(m[r], 2, 32));
        m[r] = fmaxf(m[r], __shfl_xor(m[r], 4, 32));
        m[r] = fmaxf(m[r], __shfl_xor(m[r], 8, 32));
    }

    // ---- exp pass + row sums --------------------------------------------
    float sum[8];
#pragma unroll
    for (int r = 0; r < 8; ++r) sum[r] = 0.f;

    for (int jt = 0; jt < 32; ++jt) {
        const int col = jt * 16 + l16;
#pragma unroll
        for (int r = 0; r < 8; ++r) {
            const int row = r + half * 8;
            float e = __expf(sc[row][col] - m[r]);
            sc[row][col] = e;
            sum[r] += e;
        }
    }

#pragma unroll
    for (int r = 0; r < 8; ++r) {
        sum[r] += __shfl_xor(sum[r], 1, 32);
        sum[r] += __shfl_xor(sum[r], 2, 32);
        sum[r] += __shfl_xor(sum[r], 4, 32);
        sum[r] += __shfl_xor(sum[r], 8, 32);
    }
    float inv[8];
#pragma unroll
    for (int r = 0; r < 8; ++r) inv[r] = 1.0f / sum[r];

    // broadcast 1/rowsum for all 16 rows (lane 0 holds rows 0-7, lane 16: 8-15)
    if (lane == 0) {
#pragma unroll
        for (int r = 0; r < 8; ++r) rowinv[r] = inv[r];
    } else if (lane == 16) {
#pragma unroll
        for (int r = 0; r < 8; ++r) rowinv[r + 8] = inv[r];
    }
    __syncthreads();

    // ---- write normalized attn: one contiguous 2KB row per 4 b128 stores ----
    // lane covers cols 4*lane + 128*t  -> 32 lanes x 16B x 4 = full 2KB row
    for (int row = 0; row < 16; ++row) {
        const float riv = rowinv[row];
        float* arow = attn + ((size_t)bh * LQ + q0 + row) * LK;
#pragma unroll
        for (int t = 0; t < 4; ++t) {
            const int c = t * 128 + lane * 4;
            v4f e = *(const v4f*)&sc[row][c];      // ds_load_b128 (16B aligned)
            v4f o;
#pragma unroll
            for (int i = 0; i < 4; ++i) o[i] = e[i] * riv;
            *(v4f*)(arow + c) = o;                 // global_store_b128
        }
    }

    // ---- GEMM2: out = attn @ Vsum  (K=512 in 16 chunks, N=64 in 4 tiles) --
    v8f acc[4];
#pragma unroll
    for (int nt = 0; nt < 4; ++nt) acc[nt] = (v8f){};

    const _Float16* vb = vsumT + (size_t)bh * Dc * LK;
    for (int kc = 0; kc < 16; ++kc) {
        // A from LDS (unnormalized exp): lane M = l16; 4x ds_load_b128
        const float* srow = &sc[l16][kc * 32 + half * 16];
        v4f f0 = *(const v4f*)(srow);              // K = kc*32 + half*16 + 0..3
        v4f f1 = *(const v4f*)(srow + 4);          //                     + 4..7
        v4f f2 = *(const v4f*)(srow + 8);          //                     + 8..11
        v4f f3 = *(const v4f*)(srow + 12);         //                     + 12..15
        v16h a;
#pragma unroll
        for (int e = 0; e < 4; ++e) {
            a[e]      = (_Float16)f0[e];
            a[e + 4]  = (_Float16)f1[e];
            a[e + 8]  = (_Float16)f2[e];
            a[e + 12] = (_Float16)f3[e];
        }
#pragma unroll
        for (int nt = 0; nt < 4; ++nt) {
            const int n = nt * 16 + l16;           // output d index
            v16h bb = *(const v16h*)(vb + (size_t)n * LK + kc * 32 + half * 16);
            acc[nt] = __builtin_amdgcn_wmma_f32_16x16x32_f16(false, a, false, bb,
                                                             (short)0, acc[nt],
                                                             false, false);
        }
    }

    // ---- scale by 1/rowsum and store output ------------------------------
#pragma unroll
    for (int nt = 0; nt < 4; ++nt) {
#pragma unroll
        for (int r = 0; r < 8; ++r) {
            const int row = r + half * 8;
            out[((size_t)bh * LQ + q0 + row) * DV + nt * 16 + l16] =
                acc[nt][r] * inv[r];
        }
    }
}

// ---------------------------------------------------------------------------
extern "C" void kernel_launch(void* const* d_in, const int* in_sizes, int n_in,
                              void* d_out, int out_size, void* d_ws, size_t ws_size,
                              hipStream_t stream)
{
    const float*         q     = (const float*)d_in[0];
    const float*         k     = (const float*)d_in[1];
    const float*         v     = (const float*)d_in[2];
    const float*         omega = (const float*)d_in[3];
    const unsigned char* mask  = (const unsigned char*)d_in[4]; // jnp bool = 1B

    float* out  = (float*)d_out;                       // (BH, LQ, 64)
    float* attn = out + (size_t)BH * LQ * DV;          // (BH, LQ, LK)

    _Float16* ksum  = (_Float16*)d_ws;                         // 4 MB
    _Float16* vsumT = ksum + (size_t)BH * LK * Dc;             // 4 MB

    {
        const int total = BH * LK * Dc;                // 2,097,152
        dim3 grid((total + 255) / 256), block(256);
        kv_reduce_kernel<<<grid, block, 0, stream>>>(k, v, ksum, vsumT);
    }
    {
        dim3 grid(BH * (LQ / 16)), block(32);          // 2048 waves
        attn_wmma_kernel<<<grid, block, 0, stream>>>(q, omega, mask,
                                                     ksum, vsumT, out, attn);
    }
}
